// FINN_DiffAD2ss_20864951124289
// MI455X (gfx1250) — matrix operand order repaired and checked by reading, not compiled, not run on hardware
//
#include <hip/hip_runtime.h>
#include <math.h>

typedef __attribute__((ext_vector_type(16))) _Float16 v16h;
typedef __attribute__((ext_vector_type(8)))  float    v8f;

#define NXP     65536
#define TSTEPS  200
#define HIDN    32
#define WAVES   8
#define PTSW    32              // points per wave: two 16-column WMMA tiles
#define PTSB    (WAVES * PTSW)  // 256 points per block
#define NBLK    (NXP / PTSB)    // 256 blocks

#define LOG2E   1.44269504088896340736f

// Branch-free activations on the hardware transcendental pipe.
__device__ __forceinline__ float fast_tanh(float x) {
    float e = __builtin_amdgcn_exp2f(x * (2.0f * LOG2E));
    return 1.0f - 2.0f * __builtin_amdgcn_rcpf(e + 1.0f);
}
__device__ __forceinline__ float fast_sigmoid(float x) {
    float e = __builtin_amdgcn_exp2f(-x * LOG2E);
    return __builtin_amdgcn_rcpf(1.0f + e);
}
__device__ __forceinline__ float fast_pow(float x, float y) {
    return __builtin_amdgcn_exp2f(y * __builtin_amdgcn_logf(x));
}

// Wave-uniform weight read: both addresses are uniform (constant index), so
// the compiler emits scalar s_loads (SGPRs, KMcnt) + one v_cndmask on `half`
// instead of a per-lane VMEM load.
#define USEL(P, LO, HI) ((half) ? (P)[HI] : (P)[LO])

// ---------------------------------------------------------------------------
// One-shot prep: f16 A-matrix fragments of W1^T / W2^T in this file's contiguous-k per-lane
// register layout, stored per-lane contiguous (32 B -> two b128 loads).
//   A is 16x32 (MxK): lane holds row M = lane%16; element e -> K:
//     K = (lane>>4)*16 + e
//   A[M,K] = W[K, jg], jg = tile*16 + lane%16   (h_out = W^T h_in)
// ---------------------------------------------------------------------------
__global__ __launch_bounds__(32)
void finn_prep(const float* __restrict__ W1, const float* __restrict__ W2,
               _Float16* __restrict__ frag)
{
    const int lane = threadIdx.x & 31;
    const int half = lane >> 4;
    const int lp   = lane & 15;
    v16h a1lo, a1hi, a2lo, a2hi;
    #pragma unroll
    for (int e = 0; e < 16; ++e) {
        const int K = (half << 4) + e;
        a1lo[e] = (_Float16)W1[K * HIDN + lp];
        a1hi[e] = (_Float16)W1[K * HIDN + 16 + lp];
        a2lo[e] = (_Float16)W2[K * HIDN + lp];
        a2hi[e] = (_Float16)W2[K * HIDN + 16 + lp];
    }
    v16h* out = (v16h*)frag;
    out[0 * 32 + lane] = a1lo;
    out[1 * 32 + lane] = a1hi;
    out[2 * 32 + lane] = a2lo;
    out[3 * 32 + lane] = a2hi;
}

// One hidden layer for one 16-point tile. The bias is pre-staged in the WMMA
// C-operand layout (lane holds row M = half*8+r), so D = W^T*H + bias comes
// straight out of the matrix pipe. LDS tile is strictly per-wave and the
// CDNA5 DS pipe is in-order per wave -> compiler fence instead of s_barrier.
__device__ __forceinline__ v16h mlp_layer(
    const v16h alo, const v16h ahi, const v16h bh,
    const v8f cblo, const v8f cbhi,
    _Float16 (* __restrict__ tile)[HIDN],   // [16][32] per-wave slice
    const int lp, const int half)
{
    const v8f dlo = __builtin_amdgcn_wmma_f32_16x16x32_f16(
        false, alo, false, bh, (short)0, cblo, false, false);
    const v8f dhi = __builtin_amdgcn_wmma_f32_16x16x32_f16(
        false, ahi, false, bh, (short)0, cbhi, false, false);
    // D layout: lane holds M = half*8 + r, N = lp
    #pragma unroll
    for (int r = 0; r < 8; ++r) {
        const int j = (half << 3) + r;
        tile[lp][j]      = (_Float16)fast_tanh(dlo[r]);
        tile[lp][16 + j] = (_Float16)fast_tanh(dhi[r]);
    }
    __builtin_amdgcn_wave_barrier();
    v16h out;   // B layout: lane holds column N = lp, element e -> K = half*16+e
    #pragma unroll
    for (int e = 0; e < 16; ++e)
        out[e] = tile[lp][(half << 4) + e];
    __builtin_amdgcn_wave_barrier();
    return out;
}

__global__ __launch_bounds__(256)
void finn_diffad_step(
    const float* __restrict__ u_src,   // [NXP][2] current state
    float*       __restrict__ u_dst,   // [NXP][2] next state
    float*       __restrict__ traj,    // [NXP][2] trajectory slice for this step
    const _Float16* __restrict__ frag, // prepped W1^T/W2^T fragments
    const float* __restrict__ W0, const float* __restrict__ b0,   // (1,32),(32)
    const float* __restrict__ b1, const float* __restrict__ b2,   // (32),(32)
    const float* __restrict__ W3, const float* __restrict__ b3,   // (32,1),(1)
    const float* __restrict__ tArr,
    const float* __restrict__ ret_fac_p, const float* __restrict__ D_p,
    const float* __restrict__ stencil,   const float* __restrict__ BC_p,
    const float* __restrict__ f_p,       const float* __restrict__ k_d_p,
    const float* __restrict__ beta_p,    const float* __restrict__ alpha_p,
    const float* __restrict__ rho_s_p,   const float* __restrict__ n_e_p,
    const float* __restrict__ v_e_p,     const float* __restrict__ dx_p)
{
    // per-wave 2 tiles of 16(points) x 32(hidden) f16 for D->B re-layout
    __shared__ _Float16 hbuf[WAVES][2][16][HIDN];

    const int lane = threadIdx.x & 31;
    const int wave = threadIdx.x >> 5;
    const int half = lane >> 4;    // 0: lanes 0-15, 1: lanes 16-31
    const int lp   = lane & 15;    // WMMA column within a tile
    const int base = blockIdx.x * PTSB + wave * PTSW;
    const int q    = base + lane;  // point this lane owns for state/PDE

    // ---------------- scalars (uniform -> scalar loads) ----------------
    const float dt      = tArr[1] - tArr[0];
    const float ret_fac = ret_fac_p[0];
    const float Dco     = D_p[0];
    const float s0      = stencil[0], s1 = stencil[1];
    const float BC      = BC_p[0];
    const float ff      = f_p[0];
    const float k_d     = k_d_p[0];
    const float beta    = beta_p[0];
    const float alpha   = alpha_p[0];
    const float rho_s   = rho_s_p[0];
    const float n_e     = n_e_p[0];
    const float v_e     = v_e_p[0];
    const float dx      = dx_p[0];

    // ---------------- prepped A fragments (8x global_load_b128) -------
    const v16h* fb = (const v16h*)frag;
    const v16h a1lo = fb[0 * 32 + lane];
    const v16h a1hi = fb[1 * 32 + lane];
    const v16h a2lo = fb[2 * 32 + lane];
    const v16h a2hi = fb[3 * 32 + lane];

    // ---------------- state: one float2 + one cross-half shuffle ------
    const float2 uq = *(const float2*)(u_src + 2 * q);
    const float  cq = uq.x, skq = uq.y;
    const float  cp = __shfl_xor(cq, 16, 32);     // partner half's c
    const float  cA = half ? cp : cq;             // tile0 column lp value
    const float  cB = half ? cq : cp;             // tile1 column lp value

    // ---------------- layer 0 (B layout, scalar W0/b0 + cndmask) ------
    v16h bh0, bh1;
    #pragma unroll
    for (int e = 0; e < 16; ++e) {
        const float w = USEL(W0, e, 16 + e);   // k = half*16 + e
        const float b = USEL(b0, e, 16 + e);
        bh0[e] = (_Float16)fast_tanh(cA * w + b);
        bh1[e] = (_Float16)fast_tanh(cB * w + b);
    }

    // ---------------- hidden layers ------------------------------------
    _Float16 (* __restrict__ t0)[HIDN] = hbuf[wave][0];
    _Float16 (* __restrict__ t1)[HIDN] = hbuf[wave][1];

    // b1 staged in C layout: C[r] = b[(tile?16:0) + half*8 + r]
    v8f cb_lo, cb_hi;
    #pragma unroll
    for (int r = 0; r < 8; ++r) {
        cb_lo[r] = USEL(b1, r,      8 + r);
        cb_hi[r] = USEL(b1, 16 + r, 24 + r);
    }
    bh0 = mlp_layer(a1lo, a1hi, bh0, cb_lo, cb_hi, t0, lp, half);
    bh1 = mlp_layer(a1lo, a1hi, bh1, cb_lo, cb_hi, t1, lp, half);

    #pragma unroll
    for (int r = 0; r < 8; ++r) {
        cb_lo[r] = USEL(b2, r,      8 + r);
        cb_hi[r] = USEL(b2, 16 + r, 24 + r);
    }
    bh0 = mlp_layer(a2lo, a2hi, bh0, cb_lo, cb_hi, t0, lp, half);
    bh1 = mlp_layer(a2lo, a2hi, bh1, cb_lo, cb_hi, t1, lp, half);

    // ---------------- layer 3: dots + cross-half reduce ----------------
    float pa = 0.0f, pb = 0.0f;
    #pragma unroll
    for (int e = 0; e < 16; ++e) {
        const float w = USEL(W3, e, 16 + e);
        pa += w * (float)bh0[e];
        pb += w * (float)bh1[e];
    }
    const float ta  = pa + __shfl_xor(pa, 16, 32);   // tile0, column lp
    const float tb  = pb + __shfl_xor(pb, 16, 32);   // tile1, column lp
    const float tot = half ? tb : ta;                // MLP output at point q
    const float ret = fast_sigmoid(tot + b3[0]) * ret_fac;

    // ---------------- PDE update: all 32 lanes, point q ----------------
    const float cl = (q == 0)       ? BC : u_src[2 * (q - 1) + 0];
    const float cr = (q == NXP - 1) ? cq : u_src[2 * (q + 1) + 0];

    float lf = s0 * cq + s1 * cl;
    if (q == 0) lf *= 2.0f;
    const float rf   = s0 * cq + s1 * cr;
    const float diff = (Dco / (dx * dx)) * (lf + rf);
    const float adv  = (v_e / dx) * (cq - cl);
    const float dsk  = alpha * ((1.0f - ff) * k_d *
                                fast_pow(fmaxf(cq, 1e-9f), beta) - skq);
    const float dc   = ret * (diff - adv) - (rho_s / n_e) * dsk;

    const float2 un = make_float2(cq + dt * dc, skq + dt * dsk);
    *(float2*)(u_dst + 2 * q) = un;
    *(float2*)(traj  + 2 * q) = un;
}

extern "C" void kernel_launch(void* const* d_in, const int* in_sizes, int n_in,
                              void* d_out, int out_size, void* d_ws, size_t ws_size,
                              hipStream_t stream)
{
    const float* u0      = (const float*)d_in[0];
    const float* tArr    = (const float*)d_in[1];
    const float* W0      = (const float*)d_in[2];
    const float* b0      = (const float*)d_in[3];
    const float* W1      = (const float*)d_in[4];
    const float* b1      = (const float*)d_in[5];
    const float* W2      = (const float*)d_in[6];
    const float* b2      = (const float*)d_in[7];
    const float* W3      = (const float*)d_in[8];
    const float* b3      = (const float*)d_in[9];
    const float* ret_fac = (const float*)d_in[10];
    const float* Dco     = (const float*)d_in[11];
    const float* stencil = (const float*)d_in[12];
    const float* BC      = (const float*)d_in[13];
    const float* f       = (const float*)d_in[14];
    const float* k_d     = (const float*)d_in[15];
    const float* beta    = (const float*)d_in[16];
    const float* alpha   = (const float*)d_in[17];
    const float* rho_s   = (const float*)d_in[18];
    const float* n_e     = (const float*)d_in[19];
    const float* v_e     = (const float*)d_in[20];
    const float* dx      = (const float*)d_in[21];

    float*     traj = (float*)d_out;
    float*     bufA = (float*)d_ws;                         // ping (512 KB)
    float*     bufB = bufA + (size_t)NXP * 2;               // pong (512 KB)
    _Float16*  frag = (_Float16*)(bufB + (size_t)NXP * 2);  // 4 KB fragments

    finn_prep<<<1, 32, 0, stream>>>(W1, W2, frag);

    for (int s = 0; s < TSTEPS; ++s) {
        const float* src = (s == 0) ? u0 : ((s & 1) ? bufA : bufB);
        float*       dst = (s & 1) ? bufB : bufA;
        float*       tr  = traj + (size_t)s * NXP * 2;
        finn_diffad_step<<<NBLK, 256, 0, stream>>>(
            src, dst, tr, frag,
            W0, b0, b1, b2, W3, b3,
            tArr, ret_fac, Dco, stencil, BC, f, k_d, beta,
            alpha, rho_s, n_e, v_e, dx);
    }
}
